// Qwen3VLMoeTextSparseMoeBlock_86792699117786
// MI455X (gfx1250) — matrix operand order repaired and checked by reading, not compiled, not run on hardware
//
#include <hip/hip_runtime.h>
#include <hip/hip_bf16.h>
#include <stdint.h>

// ---------------- fixed problem shape (from reference setup_inputs) ---------
#define HDIM   2048   // hidden size H
#define NEXP   32     // experts E
#define IDIM   768    // intermediate I
#define TWOI   1536   // 2*I
#define TOPK   4

#define ASYNC_G2L 1   // use global_load_async_to_lds_b128 for bf16 tile copies

typedef __bf16 bf16_t;
typedef bf16_t v16bf __attribute__((ext_vector_type(16)));
typedef float  v8f   __attribute__((ext_vector_type(8)));

// ---- fp32 -> bf16 helpers (native casts -> v_cvt_pk_bf16_f32 where possible)
__device__ __forceinline__ unsigned short f2bf1(float a) {
    union { bf16_t h; unsigned short u; } un;
    un.h = (bf16_t)a;
    return un.u;
}
__device__ __forceinline__ unsigned int f2bf_pk(float a, float b) {
    union { bf16_t h[2]; unsigned int u; } un;
    un.h[0] = (bf16_t)a;
    un.h[1] = (bf16_t)b;
    return un.u;
}

__device__ __forceinline__ float fast_rcp(float x) {
#if __has_builtin(__builtin_amdgcn_rcpf)
    return __builtin_amdgcn_rcpf(x);
#else
    return 1.f / x;
#endif
}

// ---- async global(bf16) -> LDS 16B copy ------------------------------------
__device__ __forceinline__ void g2l_b128(void* lds_dst, const void* gsrc) {
#if ASYNC_G2L
    unsigned int l = (unsigned int)(size_t)lds_dst;   // low 32 bits = LDS offset
    asm volatile("global_load_async_to_lds_b128 %0, %1, off"
                 :: "v"(l), "v"(gsrc) : "memory");
#else
    *(uint4*)lds_dst = *(const uint4*)gsrc;
#endif
}
__device__ __forceinline__ void wait_async() {
#if ASYNC_G2L
    asm volatile("s_wait_asynccnt 0x0" ::: "memory");
#endif
}

// Build A fragment (16x32 bf16, MxK) for v_wmma_f32_16x16x32_bf16.
// LDS layout: a32[row*16 + kpair]  (16 u32 = 32 bf16 per row)
// lane<16 : row=lane,    K = 0..7  and 16..23
// lane>=16: row=lane-16, K = 8..15 and 24..31
__device__ __forceinline__ v16bf load_a_frag(const unsigned int* a32, int lane) {
    int row  = lane & 15;
    int half = lane >> 4;
    const uint4* ap = (const uint4*)(a32 + row * 16);
    union { uint4 q[2]; v16bf v; } u;
    u.q[0] = ap[half];       // u32 0..3  or 4..7
    u.q[1] = ap[2 + half];   // u32 8..11 or 12..15
    return u.v;
}

// Build B fragment (32x16 bf16, KxN).
// LDS layout: column-major-in-K u32: b32[col*16 + kpair] (32 bf16 per column)
// lane<16 : col=ccbase+lane,    K = 0..7  and 16..23
// lane>=16: col=ccbase+lane-16, K = 8..15 and 24..31
__device__ __forceinline__ v16bf load_b_frag(const unsigned int* b32, int ccbase, int lane) {
    int c     = ccbase + (lane & 15);
    int khalf = lane >> 4;
    const uint4* bp = (const uint4*)(b32 + c * 16);
    union { uint4 q[2]; v16bf v; } u;
    u.q[0] = bp[khalf];
    u.q[1] = bp[2 + khalf];
    return u.v;
}

__device__ __forceinline__ v8f wmma_bf16(v16bf a, v16bf b, v8f c) {
    return __builtin_amdgcn_wmma_f32_16x16x32_bf16(false, a, false, b, (short)0, c, false, false);
}

// ---------------------------------------------------------------------------
// 1) fp32 -> bf16 conversion of activations (packed, 4 per thread)
// ---------------------------------------------------------------------------
__global__ __launch_bounds__(256) void k_convert_x(const float* __restrict__ x,
                                                   uint2* __restrict__ xb64, int n4) {
    int i = blockIdx.x * 256 + threadIdx.x;   // float4 index
    if (i < n4) {
        float4 v = ((const float4*)x)[i];
        uint2 r;
        r.x = f2bf_pk(v.x, v.y);
        r.y = f2bf_pk(v.z, v.w);
        xb64[i] = r;
    }
}

// ---------------------------------------------------------------------------
// 2) Router: logits, softmax, top-4 (renormalized).  One block per token.
// ---------------------------------------------------------------------------
__global__ __launch_bounds__(256) void k_router(const float* __restrict__ x,
                                                const float* __restrict__ gw,
                                                float* __restrict__ logits_out,
                                                int*   __restrict__ top_idx,
                                                float* __restrict__ top_w) {
    int t = blockIdx.x;
    int j = threadIdx.x;
    int e = j & 31;
    int c = j >> 5;                       // 8 chunks of 256
    __shared__ float part[256];

    const float* xr = x  + (size_t)t * HDIM + c * 256;
    const float* gr = gw + (size_t)e * HDIM + c * 256;
    float s = 0.f;
    #pragma unroll 8
    for (int h = 0; h < 256; ++h) s += xr[h] * gr[h];
    part[j] = s;
    __syncthreads();

    if (j < 32) {                         // wave 0, lane e holds expert e
        float logit = 0.f;
        #pragma unroll
        for (int cc = 0; cc < 8; ++cc) logit += part[cc * 32 + e];
        logits_out[(size_t)t * NEXP + e] = logit;

        // softmax over 32 lanes
        float m = logit;
        for (int i = 16; i > 0; i >>= 1) m = fmaxf(m, __shfl_xor(m, i, 32));
        float p = __expf(logit - m);
        float sum = p;
        for (int i = 16; i > 0; i >>= 1) sum += __shfl_xor(sum, i, 32);
        p /= sum;

        // iterative top-4 with index-carrying butterfly max
        float val = p;
        float tv[TOPK]; int ti[TOPK]; float ws = 0.f;
        #pragma unroll
        for (int k = 0; k < TOPK; ++k) {
            float v = val; int vi = e;
            for (int i = 16; i > 0; i >>= 1) {
                float ov = __shfl_xor(v, i, 32);
                int   oi = __shfl_xor(vi, i, 32);
                if (ov > v || (ov == v && oi < vi)) { v = ov; vi = oi; }
            }
            if (e == vi) val = -1e30f;
            tv[k] = v; ti[k] = vi; ws += v;
        }
        if (e == 0) {
            float inv = fast_rcp(ws);
            #pragma unroll
            for (int k = 0; k < TOPK; ++k) {
                top_idx[t * TOPK + k] = ti[k];
                top_w  [t * TOPK + k] = tv[k] * inv;
            }
        }
    }
}

// ---------------------------------------------------------------------------
// 3) Deterministic per-expert token lists.  1 block, wave e = expert e.
// ---------------------------------------------------------------------------
__global__ __launch_bounds__(1024) void k_build_lists(const int* __restrict__ top_idx,
                                                      const float* __restrict__ top_w,
                                                      int*   __restrict__ list_token,
                                                      float* __restrict__ list_w,
                                                      int*   __restrict__ counts,
                                                      int*   __restrict__ offsets, int T) {
    int j = threadIdx.x, e = j >> 5, lane = j & 31;
    __shared__ int s_cnt[NEXP];
    __shared__ int s_off[NEXP];

    int cnt = 0;
    for (int t0 = 0; t0 < T; t0 += 32) {
        int t = t0 + lane;
        int found = 0;
        #pragma unroll
        for (int k = 0; k < TOPK; ++k) found |= (top_idx[t * TOPK + k] == e);
        unsigned long long m = __ballot(found);
        cnt += __popcll(m);
    }
    if (lane == 0) s_cnt[e] = cnt;
    __syncthreads();
    if (j == 0) {
        int o = 0;
        for (int i = 0; i < NEXP; ++i) { s_off[i] = o; o += s_cnt[i]; }
    }
    __syncthreads();

    int pos = s_off[e];
    for (int t0 = 0; t0 < T; t0 += 32) {
        int t = t0 + lane;
        int found = -1;
        #pragma unroll
        for (int k = 0; k < TOPK; ++k) if (top_idx[t * TOPK + k] == e) found = k;
        unsigned long long m = __ballot(found >= 0);
        int p = pos + __popcll(m & ((1ull << lane) - 1ull));
        if (found >= 0) {
            list_token[p] = t;
            list_w[p]     = top_w[t * TOPK + found];
        }
        pos += __popcll(m);
    }
    if (lane == 0) { counts[e] = cnt; offsets[e] = s_off[e]; }
}

// ---------------------------------------------------------------------------
// 4) gate_up GEMM + silu.  64 tokens x 128 h-cols per block, 8 waves.
//    Ping-pong LDS: stage k0+32 while computing k0; one barrier per K-step.
//    grid = (IDIM/128 = 6, 32 token-tiles, 32 experts)
// ---------------------------------------------------------------------------
__global__ __launch_bounds__(256) void k_gate_up(const unsigned short* __restrict__ xb,   // [T,H] bf16
                                                 const float* __restrict__ gup,           // [E,H,2I]
                                                 const int*   __restrict__ list_token,
                                                 const int*   __restrict__ offsets,
                                                 const int*   __restrict__ counts,
                                                 unsigned short* __restrict__ hbuf) {     // [4T,I] bf16
    int e = blockIdx.z;
    int cnt = counts[e];
    int tt = blockIdx.y;                 // 64-token tile
    if (tt * 64 >= cnt) return;
    int off = offsets[e];
    int c0  = blockIdx.x * 128;          // h-column base inside I

    int j = threadIdx.x, lane = j & 31, w = j >> 5;

    __shared__ int          s_tok[64];
    __shared__ unsigned int aBuf[2][64 * 16];      // 2 x 4KB
    __shared__ unsigned int bBuf[2][256 * 16];     // 2 x 16KB (128 gate + 128 up cols)

    if (j < 64) s_tok[j] = list_token[off + min(tt * 64 + j, cnt - 1)];
    __syncthreads();

    // per-thread A-copy assignment: one 16B chunk per K-step
    int arow = j >> 2, aseg = j & 3;
    const unsigned short* axrow = xb + (size_t)s_tok[arow] * HDIM + aseg * 8;
    const float* gbase = gup + (size_t)e * HDIM * TWOI;

    // stage A (async b128) + B (float2 loads -> packed bf16) for one K-step
    auto stageAB = [&](unsigned int* aDst, unsigned int* bDst, int k0) {
        g2l_b128(&aDst[j * 4], axrow + k0);
        #pragma unroll
        for (int it = 0; it < 8; ++it) {
            int idx = j + it * 256;       // 0..2047
            int cp  = idx & 127;          // column pair
            int kp  = idx >> 7;           // K pair 0..15
            int cc  = cp * 2;
            int col = c0 + cc + ((cc < 128) ? 0 : (IDIM - 128));
            const float* gp = gbase + (size_t)(k0 + 2 * kp) * TWOI + col;
            float2 lo = *(const float2*)gp;
            float2 hi = *(const float2*)(gp + TWOI);
            bDst[cc * 16 + kp]       = f2bf_pk(lo.x, hi.x);
            bDst[(cc + 1) * 16 + kp] = f2bf_pk(lo.y, hi.y);
        }
    };

    v8f accG[4] = {v8f{}, v8f{}, v8f{}, v8f{}};
    v8f accU[4] = {v8f{}, v8f{}, v8f{}, v8f{}};

    int p = 0;
    stageAB(aBuf[0], bBuf[0], 0);
    for (int k0 = 0; k0 < HDIM; k0 += 32) {
        wait_async();
        __syncthreads();
        if (k0 + 32 < HDIM) stageAB(aBuf[p ^ 1], bBuf[p ^ 1], k0 + 32);

        const unsigned int* aC = aBuf[p];
        const unsigned int* bC = bBuf[p];
        v16bf bg = load_b_frag(bC, w * 16, lane);
        v16bf bu = load_b_frag(bC, 128 + w * 16, lane);
        v16bf af[4];
        #pragma unroll
        for (int ts = 0; ts < 4; ++ts) af[ts] = load_a_frag(aC + ts * 16 * 16, lane);
        #pragma unroll
        for (int ts = 0; ts < 4; ++ts) {
            accG[ts] = wmma_bf16(af[ts], bg, accG[ts]);
            accU[ts] = wmma_bf16(af[ts], bu, accU[ts]);
        }
        p ^= 1;
    }

    // epilogue: h = up * silu(gate), write bf16
    int n  = lane & 15;
    int mh = (lane >> 4) * 8;
    int col = c0 + w * 16 + n;
    #pragma unroll
    for (int ts = 0; ts < 4; ++ts) {
        #pragma unroll
        for (int i = 0; i < 8; ++i) {
            int trow = tt * 64 + ts * 16 + mh + i;
            if (trow < cnt) {
                float g = accG[ts][i], u = accU[ts][i];
                float hval = u * g * fast_rcp(1.f + __expf(-g));
                hbuf[(size_t)(off + trow) * IDIM + col] = f2bf1(hval);
            }
        }
    }
}

// ---------------------------------------------------------------------------
// 5) down GEMM + weighted scatter-add.  64 tokens x 128 out-cols per block.
//    Ping-pong LDS, one barrier per K-step.
//    grid = (HDIM/128 = 16, 32 token-tiles, 32 experts)
// ---------------------------------------------------------------------------
__global__ __launch_bounds__(256) void k_down(const unsigned short* __restrict__ hbuf,  // [4T,I] bf16
                                              const float* __restrict__ dwn,            // [E,I,H]
                                              const int*   __restrict__ list_token,
                                              const float* __restrict__ list_w,
                                              const int*   __restrict__ offsets,
                                              const int*   __restrict__ counts,
                                              float* __restrict__ out) {                // [T,H]
    int e = blockIdx.z;
    int cnt = counts[e];
    int tt = blockIdx.y;
    if (tt * 64 >= cnt) return;
    int off = offsets[e];
    int c0  = blockIdx.x * 128;

    int j = threadIdx.x, lane = j & 31, w = j >> 5;

    __shared__ unsigned int aBuf[2][64 * 16];     // 2 x 4KB
    __shared__ unsigned int bBuf[2][128 * 16];    // 2 x 8KB

    // per-thread A-copy assignment (rows are contiguous slots, no gather)
    int arow = j >> 2, aseg = j & 3;
    int aslot = off + min(tt * 64 + arow, cnt - 1);
    const unsigned short* ahrow = hbuf + (size_t)aslot * IDIM + aseg * 8;
    const float* dbase = dwn + (size_t)e * IDIM * HDIM;

    auto stageAB = [&](unsigned int* aDst, unsigned int* bDst, int k0) {
        g2l_b128(&aDst[j * 4], ahrow + k0);
        #pragma unroll
        for (int it = 0; it < 4; ++it) {
            int idx = j + it * 256;       // 0..1023
            int cp  = idx & 63;
            int kp  = idx >> 6;           // 0..15
            int cc  = cp * 2;
            const float* gp = dbase + (size_t)(k0 + 2 * kp) * HDIM + (c0 + cc);
            float2 lo = *(const float2*)gp;
            float2 hi = *(const float2*)(gp + HDIM);
            bDst[cc * 16 + kp]       = f2bf_pk(lo.x, hi.x);
            bDst[(cc + 1) * 16 + kp] = f2bf_pk(lo.y, hi.y);
        }
    };

    v8f acc[4] = {v8f{}, v8f{}, v8f{}, v8f{}};

    int p = 0;
    stageAB(aBuf[0], bBuf[0], 0);
    for (int k0 = 0; k0 < IDIM; k0 += 32) {
        wait_async();
        __syncthreads();
        if (k0 + 32 < IDIM) stageAB(aBuf[p ^ 1], bBuf[p ^ 1], k0 + 32);

        const unsigned int* aC = aBuf[p];
        const unsigned int* bC = bBuf[p];
        v16bf b = load_b_frag(bC, w * 16, lane);
        v16bf af[4];
        #pragma unroll
        for (int ts = 0; ts < 4; ++ts) af[ts] = load_a_frag(aC + ts * 16 * 16, lane);
        #pragma unroll
        for (int ts = 0; ts < 4; ++ts) acc[ts] = wmma_bf16(af[ts], b, acc[ts]);
        p ^= 1;
    }

    int n  = lane & 15;
    int mh = (lane >> 4) * 8;
    int col = c0 + w * 16 + n;
    #pragma unroll
    for (int ts = 0; ts < 4; ++ts) {
        #pragma unroll
        for (int i = 0; i < 8; ++i) {
            int trow = tt * 64 + ts * 16 + mh + i;
            if (trow < cnt) {
                int   tok = list_token[off + trow];
                float wt  = list_w[off + trow];
                __hip_atomic_fetch_add(&out[(size_t)tok * HDIM + col], acc[ts][i] * wt,
                                       __ATOMIC_RELAXED, __HIP_MEMORY_SCOPE_AGENT);
            }
        }
    }
}

// ---------------------------------------------------------------------------
extern "C" void kernel_launch(void* const* d_in, const int* in_sizes, int n_in,
                              void* d_out, int out_size, void* d_ws, size_t ws_size,
                              hipStream_t stream) {
    const float* x   = (const float*)d_in[0];   // [1,T,H]
    const float* gw  = (const float*)d_in[1];   // [E,H]
    const float* gup = (const float*)d_in[2];   // [E,H,2I]
    const float* dwn = (const float*)d_in[3];   // [E,I,H]

    const int T = in_sizes[0] / HDIM;           // 2048

    float* out    = (float*)d_out;              // [T,H]
    float* logits = out + (size_t)T * HDIM;     // [T,E]

    // workspace carving (256B aligned)
    char* p = (char*)d_ws;
    auto carve = [&](size_t bytes) { char* q = p; p += (bytes + 255) & ~(size_t)255; return q; };
    unsigned short* xb         = (unsigned short*)carve((size_t)T * HDIM * 2);
    unsigned short* hb         = (unsigned short*)carve((size_t)T * TOPK * IDIM * 2);
    int*            top_idx    = (int*)  carve((size_t)T * TOPK * 4);
    float*          top_w      = (float*)carve((size_t)T * TOPK * 4);
    int*            list_token = (int*)  carve((size_t)T * TOPK * 4);
    float*          list_w     = (float*)carve((size_t)T * TOPK * 4);
    int*            counts     = (int*)  carve(NEXP * 4);
    int*            offsets    = (int*)  carve(NEXP * 4);

    hipMemsetAsync(d_out, 0, (size_t)T * HDIM * sizeof(float), stream);

    k_convert_x<<<(T * HDIM / 4 + 255) / 256, 256, 0, stream>>>(x, (uint2*)xb, T * HDIM / 4);
    k_router<<<T, 256, 0, stream>>>(x, gw, logits, top_idx, top_w);
    k_build_lists<<<1, 1024, 0, stream>>>(top_idx, top_w, list_token, list_w, counts, offsets, T);

    dim3 gridGU(IDIM / 128, (T + 63) / 64, NEXP);     // (6, 32, 32)
    k_gate_up<<<gridGU, 256, 0, stream>>>(xb, gup, list_token, offsets, counts, hb);

    dim3 gridDN(HDIM / 128, (T + 63) / 64, NEXP);     // (16, 32, 32)
    k_down<<<gridDN, 256, 0, stream>>>(hb, dwn, list_token, list_w, offsets, counts, out);
}
